// myGraphDecoder_10514079941372
// MI455X (gfx1250) — matrix operand order repaired and checked by reading, not compiled, not run on hardware
//
#include <hip/hip_runtime.h>
#include <hip/hip_bf16.h>
#include <stdint.h>

#define NN 50000
#define EE 600000
#define FF 128
#define HH 256
#define OO 128

typedef __attribute__((ext_vector_type(16))) __bf16 v16bf;
typedef __attribute__((ext_vector_type(8)))  float  f32x8;

union ABu { v16bf v; uint32_t u[8]; };

__device__ __forceinline__ uint32_t pk_bf16(float a, float b) {
  uint32_t ua = __float_as_uint(a), ub = __float_as_uint(b);
  uint32_t ra = (ua + 0x7FFFu + ((ua >> 16) & 1u)) >> 16;
  uint32_t rb = (ub + 0x7FFFu + ((ub >> 16) & 1u)) >> 16;
  return (ra & 0xFFFFu) | (rb << 16);
}

__device__ __forceinline__ uint16_t cvt1_bf16(float a) {
  uint32_t u = __float_as_uint(a);
  return (uint16_t)((u + 0x7FFFu + ((u >> 16) & 1u)) >> 16);
}

// ---------------------------------------------------------------- utility
__global__ void zero_f32(float* __restrict__ p, int n) {
  int i = blockIdx.x * blockDim.x + threadIdx.x;
  if (i < n) p[i] = 0.0f;
}

__global__ void cvt_bf16_kernel(const float* __restrict__ s, uint16_t* __restrict__ d, int n) {
  int i = blockIdx.x * blockDim.x + threadIdx.x;
  if (i < n) d[i] = cvt1_bf16(s[i]);
}

// ------------------------------------------------------ edge scatter (mean agg)
// One wave (32 lanes) per edge: 4 coalesced global_atomic_add_f32 per lane.
__global__ __launch_bounds__(256) void scatter_kernel(
    const float* __restrict__ x, const long long* __restrict__ ei,
    float* __restrict__ summed, float* __restrict__ deg, int E)
{
  long long t = (long long)blockIdx.x * blockDim.x + threadIdx.x;
  int e    = (int)(t >> 5);
  int lane = (int)(t & 31);
  if (e >= E) return;
  int src = (int)ei[e];
  int dst = (int)ei[(size_t)E + e];
  if (lane == 0) atomicAdd(&deg[dst], 1.0f);
  const float* xs = x + (size_t)src * FF;
  float* sd = summed + (size_t)dst * FF;
#pragma unroll
  for (int i = 0; i < 4; ++i) {
    int f = lane + 32 * i;                 // 128B-coalesced across the wave
    atomicAdd(&sd[f], xs[f]);
  }
}

// ------------------------------------------------------ B-tile loader
// Dense 16-bit B (32x16): VGPR j, lanes 0-15: K={2j,2j+1}(j<4), {16+2(j-4),+1}(j>=4) (N=lane);
// lanes 16-31: K shifted by +8 within each half (N=lane-16). B[k][n] = W[n][k] (row-major W).
template<int LD>
__device__ __forceinline__ v16bf loadB(const uint16_t* __restrict__ W,
                                       int nb, int kb, int lane) {
  ABu b;
  const uint32_t* p = (const uint32_t*)(W + (size_t)(nb + (lane & 15)) * LD
                                          + kb + (lane >> 4) * 8);
#pragma unroll
  for (int v = 0; v < 8; ++v) b.u[v] = p[(v & 3) + (v >> 2) * 8];
  return b.v;
}

// ------------------------------------------------------ fused SAGE + MLP (WMMA)
__global__ __launch_bounds__(128) void sage_fused_wmma(
    const float* __restrict__ x, const float* __restrict__ summed,
    const float* __restrict__ deg,
    const uint16_t* __restrict__ Wl, const uint16_t* __restrict__ Wr,
    const uint16_t* __restrict__ W1,
    const float* __restrict__ bl, const float* __restrict__ b1,
    float* __restrict__ out, int ntiles)
{
  __shared__ __align__(16) uint16_t hbuf[4][16][HH];   // per-wave 8KB h staging
  const int wave = threadIdx.x >> 5;
  const int lane = threadIdx.x & 31;
  const int tile = blockIdx.x * 4 + wave;
  if (tile >= ntiles) return;                // wave-uniform: EXEC stays all-1s
  const int row = lane & 15;                 // A-row / C-column within tile
  const int kh  = lane >> 4;                 // K-half select for A layout
  const int node = tile * 16 + row;

  // ---- Assemble A operands (16x32 bf16 tiles, 4 K-steps) for mean and x ----
  // 16-bit A layout: lanes 0-15 hold K in {2v,2v+1}(v<4) and {16+2v'}(v>=4);
  // lanes 16-31 shift K by +8 within each half.
  ABu xA[4], mA[4];
  const float rdeg = 1.0f / fmaxf(deg[node], 1.0f);
  const float* xrow = x      + (size_t)node * FF;
  const float* mrow = summed + (size_t)node * FF;
#pragma unroll
  for (int ks = 0; ks < 4; ++ks) {
#pragma unroll
    for (int v = 0; v < 8; ++v) {
      const int K = ks * 32 + ((v & 4) ? 16 : 0) + kh * 8 + (v & 3) * 2;
      xA[ks].u[v] = pk_bf16(xrow[K], xrow[K + 1]);
      mA[ks].u[v] = pk_bf16(mrow[K] * rdeg, mrow[K + 1] * rdeg);
    }
  }

  // ---- GEMM1: h = relu(mean@Wl^T + x@Wr^T + bl)  -> LDS as bf16 ----
  for (int nt = 0; nt < 16; ++nt) {
    const int nb = nt * 16;
    const float bias = bl[nb + row];
    f32x8 c = {bias, bias, bias, bias, bias, bias, bias, bias};
#pragma unroll
    for (int ks = 0; ks < 4; ++ks) {
      c = __builtin_amdgcn_wmma_f32_16x16x32_bf16(false, mA[ks].v, false,
            loadB<FF>(Wl, nb, ks * 32, lane), (short)0, c, false, false);
      c = __builtin_amdgcn_wmma_f32_16x16x32_bf16(false, xA[ks].v, false,
            loadB<FF>(Wr, nb, ks * 32, lane), (short)0, c, false, false);
    }
#pragma unroll
    for (int r = 0; r < 8; ++r) {
      // C/D layout: VGPR r -> M = r (lanes 0-15) / 8+r (lanes 16-31), N = lane&15
      hbuf[wave][r + kh * 8][nb + row] = cvt1_bf16(fmaxf(c[r], 0.0f));
    }
  }

  // ---- Reload h-tile from LDS as A operands (8 K-steps of 32) ----
  ABu hA[8];
#pragma unroll
  for (int ks = 0; ks < 8; ++ks) {
#pragma unroll
    for (int v = 0; v < 8; ++v) {
      const int K = ks * 32 + ((v & 4) ? 16 : 0) + kh * 8 + (v & 3) * 2;
      hA[ks].u[v] = *(const uint32_t*)&hbuf[wave][row][K];
    }
  }

  // ---- GEMM2: out = h@W1^T + b1 ----
  for (int ot = 0; ot < 8; ++ot) {
    const int ob = ot * 16;
    const float bias = b1[ob + row];
    f32x8 c = {bias, bias, bias, bias, bias, bias, bias, bias};
#pragma unroll
    for (int ks = 0; ks < 8; ++ks) {
      c = __builtin_amdgcn_wmma_f32_16x16x32_bf16(false, hA[ks].v, false,
            loadB<HH>(W1, ob, ks * 32, lane), (short)0, c, false, false);
    }
#pragma unroll
    for (int r = 0; r < 8; ++r) {
      out[(size_t)(tile * 16 + r + kh * 8) * OO + ob + row] = c[r];
    }
  }
}

// ---------------------------------------------------------------- launch
extern "C" void kernel_launch(void* const* d_in, const int* in_sizes, int n_in,
                              void* d_out, int out_size, void* d_ws, size_t ws_size,
                              hipStream_t stream) {
  (void)in_sizes; (void)n_in; (void)out_size; (void)ws_size;
  const float*     x    = (const float*)d_in[0];
  const long long* ei   = (const long long*)d_in[1];
  const float*     Wl_f = (const float*)d_in[2];
  const float*     bl   = (const float*)d_in[3];
  const float*     Wr_f = (const float*)d_in[4];
  const float*     W1_f = (const float*)d_in[5];
  const float*     b1   = (const float*)d_in[6];
  float*           out  = (float*)d_out;

  // workspace layout (~26 MB): deg | summed | Wl_bf16 | Wr_bf16 | W1_bf16
  char* ws = (char*)d_ws;
  size_t off = 0;
  float* deg = (float*)(ws + off);       off += (((size_t)NN * 4) + 255) & ~(size_t)255;
  float* summed = (float*)(ws + off);    off += (size_t)NN * FF * 4;
  uint16_t* Wl = (uint16_t*)(ws + off);  off += (size_t)HH * FF * 2;
  uint16_t* Wr = (uint16_t*)(ws + off);  off += (size_t)HH * FF * 2;
  uint16_t* W1 = (uint16_t*)(ws + off);  off += (size_t)OO * HH * 2;

  // 1) zero accumulators
  zero_f32<<<(NN + 255) / 256, 256, 0, stream>>>(deg, NN);
  zero_f32<<<(NN * FF + 255) / 256, 256, 0, stream>>>(summed, NN * FF);

  // 2) weights -> bf16 (tiny; L2-resident, reused by every node tile)
  cvt_bf16_kernel<<<(HH * FF + 255) / 256, 256, 0, stream>>>(Wl_f, Wl, HH * FF);
  cvt_bf16_kernel<<<(HH * FF + 255) / 256, 256, 0, stream>>>(Wr_f, Wr, HH * FF);
  cvt_bf16_kernel<<<(OO * HH + 255) / 256, 256, 0, stream>>>(W1_f, W1, OO * HH);

  // 3) edge scatter (mean aggregation numerator + degree) — L2 atomics
  {
    long long threads = (long long)EE * 32;
    int blocks = (int)((threads + 255) / 256);
    scatter_kernel<<<blocks, 256, 0, stream>>>(x, ei, summed, deg, EE);
  }

  // 4) fused WMMA: mean/deg -> GEMM1+bias+relu -> GEMM2+bias -> out
  {
    const int ntiles = NN / 16;                 // 3125
    const int blocks = (ntiles + 3) / 4;        // 4 waves per block
    sage_fused_wmma<<<blocks, 128, 0, stream>>>(x, summed, deg, Wl, Wr, W1,
                                                bl, b1, out, ntiles);
  }
}
